// Attn_23158463660064
// MI455X (gfx1250) — compile-verified, matrix operand order checked by reading, not run on hardware
//
#include <hip/hip_runtime.h>
#include <hip/hip_bf16.h>

typedef _Float16 v8h  __attribute__((ext_vector_type(8)));
typedef _Float16 v16h __attribute__((ext_vector_type(16)));
typedef float    v8f  __attribute__((ext_vector_type(8)));

typedef __attribute__((address_space(3))) char lds_char;

#define DEV __device__ __forceinline__

// Async global->LDS copy of 16 bytes per lane (GLOBAL_LOAD_ASYNC_TO_LDS_B128,
// tracked by ASYNCcnt).  VDST = LDS byte address, VADDR = 64-bit global addr.
DEV void async_g2l_b128(const void* gsrc, void* ldst) {
  unsigned int la = (unsigned int)(unsigned long long)(lds_char*)ldst;
  unsigned long long ga = (unsigned long long)gsrc;
  asm volatile("global_load_async_to_lds_b128 %0, %1, off"
               :: "v"(la), "v"(ga) : "memory");
}

DEV void wait_asynccnt0() {
  asm volatile("s_wait_asynccnt 0" ::: "memory");
}

// Butterfly reductions over the 16-lane column group using DPP16 (pure VALU,
// no LDS round trips).  Steps: xor1 (quad_perm [1,0,3,2]), xor2 (quad_perm
// [2,3,0,1]), row_half_mirror (== xor4 once quads are uniform), row_mirror
// (== xor8 once octets are uniform).
#define DPP_XCHG(x, ctrl)                                                     \
  __int_as_float(__builtin_amdgcn_mov_dpp(__float_as_int(x), (ctrl), 0xF, 0xF, true))

DEV float dpp_max16(float x) {
  x = fmaxf(x, DPP_XCHG(x, 0xB1));   // quad_perm [1,0,3,2]
  x = fmaxf(x, DPP_XCHG(x, 0x4E));   // quad_perm [2,3,0,1]
  x = fmaxf(x, DPP_XCHG(x, 0x141));  // row_half_mirror
  x = fmaxf(x, DPP_XCHG(x, 0x140));  // row_mirror
  return x;
}

DEV float dpp_sum16(float x) {
  x += DPP_XCHG(x, 0xB1);
  x += DPP_XCHG(x, 0x4E);
  x += DPP_XCHG(x, 0x141);
  x += DPP_XCHG(x, 0x140);
  return x;
}

// Build a 16-half WMMA fragment for the CDNA5 16-bit A/B layout.
// lane l holds (row/col) = l%16; halves 0..7 -> K = g..g+7, halves 8..15 -> K = 16+g..16+g+7,
// where g = (l/16)*8.  p points at the start of the 32-wide K window for this lane's row.
DEV v16h frag_from(const _Float16* p, int g) {
  v8h lo = *reinterpret_cast<const v8h*>(p + g);
  v8h hi = *reinterpret_cast<const v8h*>(p + g + 16);
  v16h r;
#pragma unroll
  for (int i = 0; i < 8; ++i) { r[i] = lo[i]; r[i + 8] = hi[i]; }
  return r;
}

DEV v8f wmma16(v16h a, v16h b, v8f c) {
  return __builtin_amdgcn_wmma_f32_16x16x32_f16(false, a, false, b, (short)0, c,
                                                false, false);
}

__global__ void cvt_f32_f16(const float* __restrict__ in, _Float16* __restrict__ out, int n) {
  int i = blockIdx.x * 256 + threadIdx.x;
  if (i < n) out[i] = (_Float16)in[i];
}

// ---------------------------------------------------------------------------
// Tiled WMMA GEMM: C[M,N] = A[M,K] * B[K,N] + bias, f16 in, f32 accumulate.
// Block tile 128x64, 8 waves (4 along M, 2 along N), wave tile 32x32, BK=32.
// A tile staged global->LDS with async-to-LDS (ASYNCcnt).
// MODE 0: QKV epilogue -> scatter f16 q[B,H,S,d], k[B,H,S,d], v^T[B,H,d,S]
// MODE 1: proj epilogue -> f32 out[M,768]
// ---------------------------------------------------------------------------
template <int MODE>
__global__ __launch_bounds__(256) void gemm_wmma(
    const _Float16* __restrict__ A, const _Float16* __restrict__ Bm,
    const float* __restrict__ bias, int K, int N,
    _Float16* __restrict__ qh, _Float16* __restrict__ kh,
    _Float16* __restrict__ vhT, float* __restrict__ out) {
  __shared__ __attribute__((aligned(16))) _Float16 As[128 * 40];  // [row][k], pad 40
  __shared__ __attribute__((aligned(16))) _Float16 Bs[64 * 40];   // [n][k] transposed, pad 40

  const int tid  = threadIdx.x;
  const int wave = tid >> 5;
  const int lane = tid & 31;
  const int lm = lane & 15, lg = lane >> 4, g = lg * 8;
  const int wm = wave & 3, wn = wave >> 2;
  const int m0 = blockIdx.x * 128;
  const int n0 = blockIdx.y * 64;

  v8f acc[2][2] = {};

  for (int kk = 0; kk < K; kk += 32) {
    // Stage A tile: 128x32 halves, 16B per thread per part, async to LDS.
#pragma unroll
    for (int part = 0; part < 2; ++part) {
      int idx = part * 256 + tid;
      int row = idx >> 2, seg = idx & 3;
      const _Float16* src = A + (size_t)(m0 + row) * K + kk + seg * 8;
      _Float16* dst = As + row * 40 + seg * 8;
      async_g2l_b128(src, dst);
    }
    // Stage B tile transposed: 32x64 halves -> Bs[n][k].
    {
      int krow = tid >> 3, nseg = tid & 7;
      v8h bv = *reinterpret_cast<const v8h*>(Bm + (size_t)(kk + krow) * N + n0 + nseg * 8);
#pragma unroll
      for (int i = 0; i < 8; ++i) Bs[(nseg * 8 + i) * 40 + krow] = bv[i];
    }
    wait_asynccnt0();
    __syncthreads();

    const _Float16* a0p = As + (wm * 32 + lm) * 40;
    const _Float16* b0p = Bs + (wn * 32 + lm) * 40;
    v16h a0 = frag_from(a0p, g);
    v16h a1 = frag_from(a0p + 16 * 40, g);
    v16h b0 = frag_from(b0p, g);
    v16h b1 = frag_from(b0p + 16 * 40, g);
    acc[0][0] = wmma16(a0, b0, acc[0][0]);
    acc[0][1] = wmma16(a0, b1, acc[0][1]);
    acc[1][0] = wmma16(a1, b0, acc[1][0]);
    acc[1][1] = wmma16(a1, b1, acc[1][1]);
    __syncthreads();
  }

  // Epilogue. C layout: VGPR r, lanes 0-15 -> row r, lanes 16-31 -> row r+8; col = lane%16.
#pragma unroll
  for (int ti = 0; ti < 2; ++ti)
#pragma unroll
    for (int tj = 0; tj < 2; ++tj)
#pragma unroll
      for (int r = 0; r < 8; ++r) {
        int row = m0 + wm * 32 + ti * 16 + r + 8 * lg;
        int col = n0 + wn * 32 + tj * 16 + lm;
        float v = acc[ti][tj][r] + bias[col];
        if (MODE == 0) {
          int which = col / 768;
          int dd = col - which * 768;
          int head = dd >> 6, dc = dd & 63;
          int b = row >> 11, s = row & 2047;
          size_t bh = (size_t)(b * 12 + head);
          if (which == 0)       qh[(bh * 2048 + s) * 64 + dc] = (_Float16)v;
          else if (which == 1)  kh[(bh * 2048 + s) * 64 + dc] = (_Float16)v;
          else                  vhT[(bh * 64 + dc) * 2048 + s] = (_Float16)v;
        } else {
          out[(size_t)row * 768 + col] = v;
        }
      }
}

// ---------------------------------------------------------------------------
// One 32-key flash-attention block.  Q is pre-scaled by 1/sqrt(64), so scores
// come out of the WMMAs already scaled; only the final block needs masking.
// ---------------------------------------------------------------------------
template <bool MASKED>
DEV void attn_block(int kb, int q0, int lm, int lg, int g,
                    const _Float16* __restrict__ kp, const _Float16* __restrict__ vp,
                    v16h qa0, v16h qa1, _Float16 (*P)[32],
                    v8f (&oacc)[4], float (&m_i)[8], float (&l_i)[8]) {
  // K^T fragments: column n = key, K-dim = d (contiguous per key row).
  const _Float16* kr0 = kp + (size_t)(kb + lm) * 64;
  const _Float16* kr1 = kp + (size_t)(kb + 16 + lm) * 64;
  v16h k00 = frag_from(kr0, g);
  v16h k01 = frag_from(kr0 + 32, g);
  v16h k10 = frag_from(kr1, g);
  v16h k11 = frag_from(kr1 + 32, g);
  v8f s0{}, s1{};
  s0 = wmma16(qa0, k00, s0);
  s0 = wmma16(qa1, k01, s0);
  s1 = wmma16(qa0, k10, s1);
  s1 = wmma16(qa1, k11, s1);

  const int c0 = kb + lm, c1 = kb + 16 + lm;
#pragma unroll
  for (int r = 0; r < 8; ++r) {
    float v0 = s0[r];
    float v1 = s1[r];
    if (MASKED) {
      const int row = q0 + r + 8 * lg;
      v0 = (c0 <= row) ? v0 : -10000.0f;
      v1 = (c1 <= row) ? v1 : -10000.0f;
    }
    float mx = dpp_max16(fmaxf(v0, v1));
    float mnew  = fmaxf(m_i[r], mx);
    float alpha = __expf(m_i[r] - mnew);
    float p0 = __expf(v0 - mnew);
    float p1 = __expf(v1 - mnew);
    float rs = dpp_sum16(p0 + p1);
    l_i[r] = l_i[r] * alpha + rs;
    m_i[r] = mnew;
#pragma unroll
    for (int t = 0; t < 4; ++t) oacc[t][r] *= alpha;
    P[r + 8 * lg][lm]      = (_Float16)p0;
    P[r + 8 * lg][16 + lm] = (_Float16)p1;
  }
  asm volatile("s_wait_dscnt 0" ::: "memory");
  v16h pa = frag_from(&P[lm][0], g);  // P as A operand (16x32)
#pragma unroll
  for (int t = 0; t < 4; ++t) {
    // V^T fragment: column n = d index, K-dim = keys (contiguous in v^T row).
    v16h vb = frag_from(vp + (size_t)(t * 16 + lm) * 2048 + kb, g);
    oacc[t] = wmma16(pa, vb, oacc[t]);
  }
}

// ---------------------------------------------------------------------------
// Flash attention: 1 wave owns 16 query rows; loop 32-key blocks with online
// softmax.  scores = Q(16x64) x K^T  (4 wmma), out += P(16x32) x V (4 wmma).
// q/k: [B,H,S,64] f16,  v: [B,H,64,S] f16 (transposed), out ah: [B,S,768] f16.
// ---------------------------------------------------------------------------
__global__ __launch_bounds__(128) void attn_wmma(
    const _Float16* __restrict__ qh, const _Float16* __restrict__ kh,
    const _Float16* __restrict__ vhT, _Float16* __restrict__ ah) {
  __shared__ __attribute__((aligned(16))) _Float16 plds[4][16][32];

  const int tid  = threadIdx.x;
  const int wave = tid >> 5;
  const int lane = tid & 31;
  const int lm = lane & 15, lg = lane >> 4, g = lg * 8;
  const int bh = blockIdx.y;
  const int b = bh / 12, h = bh % 12;
  const int q0 = (blockIdx.x * 4 + wave) * 16;

  const _Float16* qp = qh + (size_t)bh * 2048 * 64;
  const _Float16* kp = kh + (size_t)bh * 2048 * 64;
  const _Float16* vp = vhT + (size_t)bh * 64 * 2048;

  const _Float16* qrow = qp + (size_t)(q0 + lm) * 64;
  v16h qa0 = frag_from(qrow, g);        // d = 0..31
  v16h qa1 = frag_from(qrow + 32, g);   // d = 32..63
  // Fold the 1/sqrt(dk) = 0.125 scale into Q (exact: power of two).
#pragma unroll
  for (int i = 0; i < 16; ++i) { qa0[i] *= (_Float16)0.125f; qa1[i] *= (_Float16)0.125f; }

  v8f oacc[4] = {};
  float m_i[8], l_i[8];
#pragma unroll
  for (int r = 0; r < 8; ++r) { m_i[r] = -1e30f; l_i[r] = 0.0f; }

  _Float16(*P)[32] = plds[wave];

  // Unmasked key blocks: kb + 31 < q0  <=>  j < q0/32.  Exactly one masked
  // block after that covers the diagonal (q0 is a multiple of 16).
  const int jfull = q0 >> 5;
  for (int j = 0; j < jfull; ++j)
    attn_block<false>(j * 32, q0, lm, lg, g, kp, vp, qa0, qa1, P, oacc, m_i, l_i);
  attn_block<true>(jfull * 32, q0, lm, lg, g, kp, vp, qa0, qa1, P, oacc, m_i, l_i);

#pragma unroll
  for (int r = 0; r < 8; ++r) {
    float inv = 1.0f / l_i[r];
    int s = q0 + r + 8 * lg;
#pragma unroll
    for (int t = 0; t < 4; ++t) {
      int col = h * 64 + t * 16 + lm;
      ah[((size_t)(b * 2048 + s)) * 768 + col] = (_Float16)(oacc[t][r] * inv);
    }
  }
}

extern "C" void kernel_launch(void* const* d_in, const int* in_sizes, int n_in,
                              void* d_out, int out_size, void* d_ws, size_t ws_size,
                              hipStream_t stream) {
  const float* x     = (const float*)d_in[0];
  const float* wqkv  = (const float*)d_in[1];
  const float* bqkv  = (const float*)d_in[2];
  const float* wproj = (const float*)d_in[3];
  const float* bproj = (const float*)d_in[4];
  float* out = (float*)d_out;

  const int Rows = 4 * 2048;  // 8192 token rows
  const int D = 768, N3 = 2304;

  _Float16* ws = (_Float16*)d_ws;
  size_t off = 0;
  _Float16* xh  = ws + off; off += (size_t)Rows * D;        // x in f16 (reused as ah)
  _Float16* wah = ws + off; off += (size_t)D * N3;          // c_attn_w f16
  _Float16* wph = ws + off; off += (size_t)D * D;           // c_proj_w f16
  _Float16* qh  = ws + off; off += (size_t)48 * 2048 * 64;  // [B,H,S,d]
  _Float16* kh  = ws + off; off += (size_t)48 * 2048 * 64;  // [B,H,S,d]
  _Float16* vhT = ws + off; off += (size_t)48 * 2048 * 64;  // [B,H,d,S]
  _Float16* ah  = xh;  // attention output reuses xh (x dead after QKV gemm)

  int n1 = Rows * D;
  cvt_f32_f16<<<(n1 + 255) / 256, 256, 0, stream>>>(x, xh, n1);
  int n2 = D * N3;
  cvt_f32_f16<<<(n2 + 255) / 256, 256, 0, stream>>>(wqkv, wah, n2);
  int n3 = D * D;
  cvt_f32_f16<<<(n3 + 255) / 256, 256, 0, stream>>>(wproj, wph, n3);

  gemm_wmma<0><<<dim3(Rows / 128, N3 / 64), 256, 0, stream>>>(
      xh, wah, bqkv, D, N3, qh, kh, vhT, nullptr);

  attn_wmma<<<dim3(2048 / 64, 48), 128, 0, stream>>>(qh, kh, vhT, ah);

  gemm_wmma<1><<<dim3(Rows / 128, D / 64), 256, 0, stream>>>(
      ah, wph, bproj, D, D, nullptr, nullptr, nullptr, out);
}
